// CrossAttnBlock_projkv_pe_rn_48077863912160
// MI455X (gfx1250) — compile-verified, matrix operand order checked by reading, not run on hardware
//
#include <hip/hip_runtime.h>
#include <math.h>

// CrossAttnBlock for MI455X (gfx1250): all GEMMs via V_WMMA_F32_16X16X32_BF16,
// f32 accumulate, f32 residual stream. Compute-bound on the 256x(196x2048 @
// 2048x1024) KV projection. GEMM A-tiles staged into LDS with async DMA
// (GLOBAL_LOAD_ASYNC_TO_LDS_B128, ASYNCcnt) and double-buffered against WMMA.

#define DIMC     512
#define HEADS    8
#define HEAD_DIM 64
#define KVDIM    2048
#define NQ       32
#define NIMG     196
#define BATCH    256
#define ATT_SCALE 0.125f   // 64^-0.5

typedef __bf16 bf16;
typedef __attribute__((ext_vector_type(16))) __bf16       bf16x16;
typedef __attribute__((ext_vector_type(8)))  float        f32x8;
typedef __attribute__((ext_vector_type(4)))  unsigned int u32x4;

union FragCast {
  bf16x16        v;
  u32x4          q[2];
  unsigned short s[16];
};

__device__ __host__ __forceinline__ bf16 f2bf(float f) {
  union { float f; unsigned u; } v; v.f = f;
  unsigned r = (v.u + 0x7FFFu + ((v.u >> 16) & 1u)) >> 16;
  unsigned short s = (unsigned short)r;
  bf16 out;
  __builtin_memcpy(&out, &s, 2);
  return out;
}

// Async DMA: copy 16 bytes global -> LDS, tracked by ASYNCcnt (ISA 15.18.3
// op 98). VDST VGPR carries the LDS byte address (low 32 bits of the generic
// shared pointer per the flat-aperture rule), VADDR the 64-bit global address.
__device__ __forceinline__ void async_copy_b128(void* lds_dst, const void* gsrc) {
  unsigned la = (unsigned)(unsigned long long)lds_dst;
  asm volatile("global_load_async_to_lds_b128 %0, %1, off"
               :: "v"(la), "v"(gsrc)
               : "memory");
}
__device__ __forceinline__ void async_wait0() {
  asm volatile("s_wait_asynccnt 0x0" ::: "memory");
}

// A/B fragment for V_WMMA_F32_16X16X32_BF16, row-major source (rows contiguous
// in k).  Lane layout (ISA 7.12.2, 16-bit A 16x32): lanes 0-15 hold row r0+l,
// k in {kc..kc+7, kc+16..kc+23}; lanes 16-31 hold the other k-halves.
__device__ __forceinline__ bf16x16 frag_ld_row(const bf16* __restrict__ base,
                                               int stride, int r0, int rmax,
                                               int kc) {
  int lane = threadIdx.x & 31;
  int l = lane & 15, hi = (lane >> 4) & 1;
  int r = r0 + l; if (r > rmax) r = rmax;
  const bf16* p = base + (size_t)r * stride + kc + hi * 8;
  FragCast f;
  f.q[0] = *reinterpret_cast<const u32x4*>(p);
  f.q[1] = *reinterpret_cast<const u32x4*>(p + 16);
  return f.v;
}

// B fragment from a column-major view: element (k, n) at base[k*kstride + n].
__device__ __forceinline__ bf16x16 frag_ld_col(const bf16* __restrict__ base,
                                               int kstride, int n0, int nmax,
                                               int kc, int kmax) {
  int lane = threadIdx.x & 31;
  int l = lane & 15, hi = (lane >> 4) & 1;
  int n = n0 + l; if (n > nmax) n = nmax;
  FragCast f;
#pragma unroll
  for (int i = 0; i < 16; ++i) {
    int k = kc + hi * 8 + i + (i >= 8 ? 8 : 0);
    if (k > kmax) k = kmax;
    f.s[i] = *reinterpret_cast<const unsigned short*>(base + (size_t)k * kstride + n);
  }
  return f.v;
}

__device__ __forceinline__ f32x8 wmma_bf16(bf16x16 a, bf16x16 b, f32x8 c) {
  return __builtin_amdgcn_wmma_f32_16x16x32_bf16(false, a, false, b,
                                                 (short)0, c, false, false);
}

// ---------------------------------------------------------------------------
// elementwise helpers
// ---------------------------------------------------------------------------
__global__ void cvt_bf16_kernel(const float* __restrict__ s, bf16* __restrict__ d, int n) {
  int i = blockIdx.x * blockDim.x + threadIdx.x;
  if (i < n) d[i] = f2bf(s[i]);
}

__global__ void add_pe_kernel(const float* __restrict__ q, const float* __restrict__ pe,
                              float* __restrict__ x, int total) {
  int i = blockIdx.x * blockDim.x + threadIdx.x;
  if (i < total) x[i] = q[i] + pe[i % (NQ * DIMC)];
}

// One wave per row: mean/var via shfl_xor reductions, bf16 output.
__global__ void ln_rows_kernel(const float* __restrict__ in, const float* __restrict__ g,
                               const float* __restrict__ bb, bf16* __restrict__ out, int C) {
  size_t row = blockIdx.x;
  const float* x = in + row * C;
  int lane = threadIdx.x;
  float s = 0.f, s2 = 0.f;
  for (int i = lane; i < C; i += 32) { float v = x[i]; s += v; s2 += v * v; }
#pragma unroll
  for (int off = 16; off; off >>= 1) {
    s  += __shfl_xor(s,  off, 32);
    s2 += __shfl_xor(s2, off, 32);
  }
  float mu = s / C;
  float var = s2 / C - mu * mu;
  float rs = rsqrtf(var + 1e-5f);
  bf16* o = out + row * C;
  for (int i = lane; i < C; i += 32)
    o[i] = f2bf((x[i] - mu) * rs * g[i] + bb[i]);
}

// ---------------------------------------------------------------------------
// Batched GEMM:  out[b] (MxN) = act(A[b] (MxK) @ W^T (NxK) + bias) (+resid)
// 8 waves / workgroup; each wave owns a 16x32 output strip (two 16x16 tiles,
// A fragment reused across both).  The block's shared 16-row A tile is staged
// 64 k at a time into LDS via async DMA, double-buffered against the WMMAs.
// Requires K % 64 == 0 and N % 256 == 0 (true for all call sites).
// ---------------------------------------------------------------------------
__global__ void gemm_bf16_kernel(const bf16* __restrict__ A, const bf16* __restrict__ W,
                                 const float* __restrict__ bias,
                                 const float* __restrict__ resid,
                                 float* __restrict__ outF, bf16* __restrict__ outH,
                                 int M, int N, int K, int act) {
  __shared__ bf16 tileA[2][16 * 64];   // 2 x 2KB double buffer

  int b    = blockIdx.z;
  int mt   = blockIdx.y;
  int tid  = threadIdx.x;
  int wave = tid >> 5;
  int lane = tid & 31;
  int l = lane & 15, hi = (lane >> 4) & 1;
  int nt0 = (blockIdx.x * 8 + wave) * 2;
  bool active = (nt0 * 16 < N);

  const bf16* Ab = A + (size_t)b * M * K;

  // cooperative staging map: threads 0..127 each move one b128 (8 bf16)
  int srow = tid >> 3;
  int skk  = (tid & 7) * 8;
  int arow = mt * 16 + srow; if (arow > M - 1) arow = M - 1;

  if (tid < 128)
    async_copy_b128(&tileA[0][srow * 64 + skk], Ab + (size_t)arow * K + skk);
  async_wait0();
  __syncthreads();

  int wr0 = nt0 * 16 + l;        if (wr0 > N - 1) wr0 = N - 1;
  int wr1 = nt0 * 16 + 16 + l;   if (wr1 > N - 1) wr1 = N - 1;
  const bf16* wpre0 = W + (size_t)wr0 * K;
  const bf16* wpre1 = W + (size_t)wr1 * K;

  f32x8 acc0 = {}, acc1 = {};
  int buf = 0;
  for (int kc0 = 0; kc0 < K; kc0 += 64) {
    bool more = (kc0 + 64) < K;
    if (more && tid < 128)   // issue next chunk's DMA before computing
      async_copy_b128(&tileA[buf ^ 1][srow * 64 + skk],
                      Ab + (size_t)arow * K + kc0 + 64 + skk);
    if (more) {
      __builtin_prefetch(wpre0 + kc0 + 64, 0, 3);   // global_prefetch_b8
      __builtin_prefetch(wpre1 + kc0 + 64, 0, 3);
    }
#pragma unroll
    for (int kc = 0; kc < 64; kc += 32) {
      bf16x16 af = frag_ld_row(&tileA[buf][0], 64, 0, 15, kc);      // ds_load
      bf16x16 b0 = frag_ld_row(W, K, nt0 * 16,       N - 1, kc0 + kc);
      bf16x16 b1 = frag_ld_row(W, K, (nt0 + 1) * 16, N - 1, kc0 + kc);
      acc0 = wmma_bf16(af, b0, acc0);
      acc1 = wmma_bf16(af, b1, acc1);
    }
    async_wait0();
    __syncthreads();
    buf ^= 1;
  }

  if (!active) return;
  int n0 = nt0 * 16 + l;
  int n1 = n0 + 16;
  float bv0 = bias ? bias[n0] : 0.f;
  float bv1 = (bias && n1 < N) ? bias[n1] : 0.f;
  size_t ob = (size_t)b * M * N;
#pragma unroll
  for (int r = 0; r < 8; ++r) {
    int m = mt * 16 + hi * 8 + r;
    if (m >= M) continue;
    float v0 = acc0[r] + bv0;
    float v1 = acc1[r] + bv1;
    if (act == 1) {
      v0 = 0.5f * v0 * (1.f + erff(v0 * 0.70710678118f));   // exact GELU
      v1 = 0.5f * v1 * (1.f + erff(v1 * 0.70710678118f));
    }
    size_t i0 = ob + (size_t)m * N + n0;
    size_t i1 = ob + (size_t)m * N + n1;
    if (resid) { v0 += resid[i0]; if (n1 < N) v1 += resid[i1]; }
    if (outF) { outF[i0] = v0; if (n1 < N) outF[i1] = v1; }
    if (outH) { outH[i0] = f2bf(v0); if (n1 < N) outH[i1] = f2bf(v1); }
  }
}

// ---------------------------------------------------------------------------
// Self-attention: one wave per (batch, head).  qkv: [B, 32, 1536] bf16 where
// col = sel*512 + h*64 + e.  scores 32x32 via WMMA, softmax in LDS, attn@V
// via WMMA with per-lane V-column gathers.
// ---------------------------------------------------------------------------
__global__ void sa_attn_kernel(const bf16* __restrict__ qkv, bf16* __restrict__ out) {
  int h = blockIdx.x, b = blockIdx.y;
  __shared__ float sc[NQ][33];
  __shared__ bf16  pb[NQ][40];
  int lane = threadIdx.x;
  int l = lane & 15, hi = (lane >> 4) & 1;
  const bf16* base = qkv + (size_t)b * NQ * 3 * DIMC + h * HEAD_DIM;

#pragma unroll
  for (int ti = 0; ti < 2; ++ti)
#pragma unroll
    for (int tj = 0; tj < 2; ++tj) {
      f32x8 acc = {};
#pragma unroll
      for (int kc = 0; kc < HEAD_DIM; kc += 32) {
        bf16x16 qa = frag_ld_row(base,        3 * DIMC, ti * 16, NQ - 1, kc);
        bf16x16 kf = frag_ld_row(base + DIMC, 3 * DIMC, tj * 16, NQ - 1, kc);
        acc = wmma_bf16(qa, kf, acc);
      }
#pragma unroll
      for (int r = 0; r < 8; ++r)
        sc[ti * 16 + hi * 8 + r][tj * 16 + l] = acc[r];
    }
  __syncthreads();

  {  // softmax: lane owns row `lane`
    float mx = -1e30f;
    for (int j = 0; j < NQ; ++j) mx = fmaxf(mx, sc[lane][j] * ATT_SCALE);
    float sum = 0.f;
    for (int j = 0; j < NQ; ++j) sum += __expf(sc[lane][j] * ATT_SCALE - mx);
    float inv = 1.f / sum;
    for (int j = 0; j < NQ; ++j)
      pb[lane][j] = f2bf(__expf(sc[lane][j] * ATT_SCALE - mx) * inv);
  }
  __syncthreads();

  const bf16* vbase = base + 2 * DIMC;
  size_t ob = (size_t)b * NQ * DIMC + h * HEAD_DIM;
#pragma unroll
  for (int ti = 0; ti < 2; ++ti)
#pragma unroll
    for (int ej = 0; ej < 4; ++ej) {
      f32x8 acc = {};
      bf16x16 pa = frag_ld_row(&pb[0][0], 40, ti * 16, NQ - 1, 0);
      bf16x16 vf = frag_ld_col(vbase + ej * 16, 3 * DIMC, 0, 15, 0, NQ - 1);
      acc = wmma_bf16(pa, vf, acc);
#pragma unroll
      for (int r = 0; r < 8; ++r) {
        int m = ti * 16 + hi * 8 + r;
        out[ob + (size_t)m * DIMC + ej * 16 + l] = f2bf(acc[r]);
      }
    }
}

// ---------------------------------------------------------------------------
// Cross-attention: one workgroup (8 waves) per batch.
// q: [B,32,512] bf16; kv: [B,196,1024] bf16 (k = cols 0..511, v = 512..1023).
// scores 32x196 (padded to 208 cols) via WMMA K=512; softmax in LDS;
// attn@V via WMMA with K padded to 224 (zero probs on the pad).
// ---------------------------------------------------------------------------
__global__ void ca_attn_kernel(const bf16* __restrict__ q, const bf16* __restrict__ kv,
                               bf16* __restrict__ out) {
  int b = blockIdx.x;
  __shared__ float sc[NQ][208];
  __shared__ bf16  pb[NQ][224];
  int wave = threadIdx.x >> 5;
  int lane = threadIdx.x & 31;
  int l = lane & 15, hi = (lane >> 4) & 1;
  const bf16* qb = q  + (size_t)b * NQ * DIMC;
  const bf16* kb = kv + (size_t)b * NIMG * 2 * DIMC;

  // scores: 2 x 13 tiles
  for (int t = wave; t < 26; t += 8) {
    int mt = t / 13, nt = t % 13;
    f32x8 acc = {};
    for (int kc = 0; kc < DIMC; kc += 32) {
      bf16x16 qa = frag_ld_row(qb, DIMC,     mt * 16, NQ - 1,   kc);
      bf16x16 kf = frag_ld_row(kb, 2 * DIMC, nt * 16, NIMG - 1, kc);
      acc = wmma_bf16(qa, kf, acc);
    }
#pragma unroll
    for (int r = 0; r < 8; ++r)
      sc[mt * 16 + hi * 8 + r][nt * 16 + l] = acc[r];
  }
  __syncthreads();

  if (threadIdx.x < NQ) {  // softmax over 196, one thread per row
    int row = threadIdx.x;
    float mx = -1e30f;
    for (int j = 0; j < NIMG; ++j) mx = fmaxf(mx, sc[row][j] * ATT_SCALE);
    float sum = 0.f;
    for (int j = 0; j < NIMG; ++j) sum += __expf(sc[row][j] * ATT_SCALE - mx);
    float inv = 1.f / sum;
    for (int j = 0; j < NIMG; ++j)
      pb[row][j] = f2bf(__expf(sc[row][j] * ATT_SCALE - mx) * inv);
    for (int j = NIMG; j < 224; ++j) pb[row][j] = f2bf(0.f);
  }
  __syncthreads();

  // out = probs(32 x 224pad) @ v(196 x 512): 2 x 32 tiles
  const bf16* vb = kb + DIMC;
  size_t ob = (size_t)b * NQ * DIMC;
  for (int t = wave; t < 64; t += 8) {
    int mt = t >> 5, nt = t & 31;
    f32x8 acc = {};
    for (int kc = 0; kc < 224; kc += 32) {
      bf16x16 pa = frag_ld_row(&pb[0][0], 224, mt * 16, NQ - 1, kc);
      bf16x16 vf = frag_ld_col(vb + nt * 16, 2 * DIMC, 0, 15, kc, NIMG - 1);
      acc = wmma_bf16(pa, vf, acc);
    }
#pragma unroll
    for (int r = 0; r < 8; ++r) {
      int m = mt * 16 + hi * 8 + r;
      out[ob + (size_t)m * DIMC + nt * 16 + l] = f2bf(acc[r]);
    }
  }
}

// ---------------------------------------------------------------------------
// host side
// ---------------------------------------------------------------------------
static void launch_gemm(const bf16* A, const bf16* W, const float* bias,
                        const float* resid, float* outF, bf16* outH,
                        int B, int M, int N, int K, int act, hipStream_t s) {
  dim3 g((N / 16 + 15) / 16, (M + 15) / 16, B);
  gemm_bf16_kernel<<<g, 256, 0, s>>>(A, W, bias, resid, outF, outH, M, N, K, act);
}

extern "C" void kernel_launch(void* const* d_in, const int* in_sizes, int n_in,
                              void* d_out, int out_size, void* d_ws, size_t ws_size,
                              hipStream_t stream) {
  (void)in_sizes; (void)n_in; (void)out_size; (void)ws_size;
  const float* query     = (const float*)d_in[0];
  const float* img_fea   = (const float*)d_in[1];
  const float* pos_embed = (const float*)d_in[2];
  const float* sa_qkv_w  = (const float*)d_in[3];
  const float* sa_proj_w = (const float*)d_in[4];
  const float* sa_proj_b = (const float*)d_in[5];
  const float* ca_q_w    = (const float*)d_in[6];
  const float* ca_kv_w   = (const float*)d_in[7];
  const float* ca_proj_w = (const float*)d_in[8];
  const float* ca_proj_b = (const float*)d_in[9];
  const float* n1_g  = (const float*)d_in[10];
  const float* n1_b  = (const float*)d_in[11];
  const float* n2q_g = (const float*)d_in[12];
  const float* n2q_b = (const float*)d_in[13];
  const float* n2kv_g = (const float*)d_in[14];
  const float* n2kv_b = (const float*)d_in[15];
  const float* nm_g  = (const float*)d_in[16];
  const float* nm_b  = (const float*)d_in[17];
  const float* fc1_w = (const float*)d_in[18];
  const float* fc1_b = (const float*)d_in[19];
  const float* fc2_w = (const float*)d_in[20];
  const float* fc2_b = (const float*)d_in[21];

  char* wsb = (char*)d_ws;
  auto carve = [&](size_t elems, size_t esz) {
    void* p = (void*)wsb;
    wsb += ((elems * esz + 255) / 256) * 256;
    return p;
  };
  const size_t XE = (size_t)BATCH * NQ * DIMC;          // 4.19M
  float* x    = (float*)carve(XE, 4);                    // residual stream f32
  bf16*  xn   = (bf16*) carve(XE, 2);                    // LN'd activations
  bf16*  kvn  = (bf16*) carve((size_t)BATCH * NIMG * KVDIM, 2);
  bf16*  qkv  = (bf16*) carve((size_t)BATCH * NQ * 3 * DIMC, 2);
  bf16*  qb   = (bf16*) carve(XE, 2);
  bf16*  kvp  = (bf16*) carve((size_t)BATCH * NIMG * 2 * DIMC, 2);
  bf16*  tmp  = (bf16*) carve(XE, 2);                    // attn-out / mlp hidden
  bf16*  wqkv = (bf16*) carve(3 * DIMC * DIMC, 2);
  bf16*  wsap = (bf16*) carve(DIMC * DIMC, 2);
  bf16*  wcaq = (bf16*) carve(DIMC * DIMC, 2);
  bf16*  wcakv= (bf16*) carve(2 * DIMC * KVDIM, 2);
  bf16*  wcap = (bf16*) carve(DIMC * DIMC, 2);
  bf16*  wfc1 = (bf16*) carve(DIMC * DIMC, 2);
  bf16*  wfc2 = (bf16*) carve(DIMC * DIMC, 2);

  auto cvt = [&](const float* s, bf16* d, int n) {
    cvt_bf16_kernel<<<(n + 255) / 256, 256, 0, stream>>>(s, d, n);
  };
  cvt(sa_qkv_w,  wqkv,  3 * DIMC * DIMC);
  cvt(sa_proj_w, wsap,  DIMC * DIMC);
  cvt(ca_q_w,    wcaq,  DIMC * DIMC);
  cvt(ca_kv_w,   wcakv, 2 * DIMC * KVDIM);
  cvt(ca_proj_w, wcap,  DIMC * DIMC);
  cvt(fc1_w,     wfc1,  DIMC * DIMC);
  cvt(fc2_w,     wfc2,  DIMC * DIMC);

  // x = query + pos_embed
  add_pe_kernel<<<(int)((XE + 255) / 256), 256, 0, stream>>>(query, pos_embed, x, (int)XE);

  // ---- self-attention branch ----
  ln_rows_kernel<<<BATCH * NQ, 32, 0, stream>>>(x, n1_g, n1_b, xn, DIMC);
  launch_gemm(xn, wqkv, nullptr, nullptr, nullptr, qkv, BATCH, NQ, 3 * DIMC, DIMC, 0, stream);
  sa_attn_kernel<<<dim3(HEADS, BATCH), 32, 0, stream>>>(qkv, tmp);
  launch_gemm(tmp, wsap, sa_proj_b, x, x, nullptr, BATCH, NQ, DIMC, DIMC, 0, stream);

  // ---- cross-attention branch ----
  ln_rows_kernel<<<BATCH * NQ, 32, 0, stream>>>(x, n2q_g, n2q_b, xn, DIMC);
  ln_rows_kernel<<<BATCH * NIMG, 32, 0, stream>>>(img_fea, n2kv_g, n2kv_b, kvn, KVDIM);
  launch_gemm(xn, wcaq, nullptr, nullptr, nullptr, qb, BATCH, NQ, DIMC, DIMC, 0, stream);
  // dominant GEMM: [196x2048] @ [2048x1024]^T per batch (822 MF x 256)
  launch_gemm(kvn, wcakv, nullptr, nullptr, nullptr, kvp, BATCH, NIMG, 2 * DIMC, KVDIM, 0, stream);
  ca_attn_kernel<<<BATCH, 256, 0, stream>>>(qb, kvp, tmp);
  launch_gemm(tmp, wcap, ca_proj_b, x, x, nullptr, BATCH, NQ, DIMC, DIMC, 0, stream);

  // ---- MLP branch ----
  ln_rows_kernel<<<BATCH * NQ, 32, 0, stream>>>(x, nm_g, nm_b, xn, DIMC);
  launch_gemm(xn, wfc1, fc1_b, nullptr, nullptr, tmp, BATCH, NQ, DIMC, DIMC, 1, stream);  // GELU
  launch_gemm(tmp, wfc2, fc2_b, x, (float*)d_out, nullptr, BATCH, NQ, DIMC, DIMC, 0, stream);
}
